// AdaptiveDeformConvND_39754217292513
// MI455X (gfx1250) — matrix operand order repaired and checked by reading, not compiled, not run on hardware
//
#include <hip/hip_runtime.h>
#include <hip/hip_bf16.h>
#include <math.h>

// Problem constants (from reference): B=4, L=4096, C=256, K=7, G=C, GC=1
#define B_   4
#define L_   4096
#define C_   256
#define K_   7
#define M_   (B_ * L_)     // 16384 rows for all GEMMs
#define NOFF (C_ * K_)     // 1792 (offset / mask GEMM width)
#define HID_ 32

typedef unsigned short u16;
typedef __attribute__((ext_vector_type(16))) __bf16 v16bf;
typedef __attribute__((ext_vector_type(8)))  float  v8f;

static __device__ __forceinline__ u16 f2bf(float f) {
  unsigned u = __float_as_uint(f);
  u += 0x7FFFu + ((u >> 16) & 1u);          // round-to-nearest-even bf16
  return (u16)(u >> 16);
}
static __device__ __forceinline__ float siluf(float x) {
  return x / (1.0f + __expf(-x));
}

// CDNA5 async memory->LDS copy (ASYNCcnt-tracked, bypasses VGPRs).
// GVS mode: LDS byte address in %0 ; global = SADDR(%2) + VADDR(%1)
static __device__ __forceinline__ void async_b128(unsigned lds_off,
                                                  const void* base,
                                                  unsigned goff) {
  asm volatile("global_load_async_to_lds_b128 %0, %1, %2"
               :: "v"(lds_off), "v"(goff), "s"(base)
               : "memory");
}

// ---------------------------------------------------------------------------
// Tiled bf16 WMMA GEMM:  C[M,N] = epi(A[M,K] @ Wt[N,K]^T + bias[N])
// A is bf16 row-major (M,K); Wt is bf16 transposed (N,K).
// Block tile 128x128, BK=32, 8 waves -> each wave a 32x64 sub-tile
// (2 x 4 v_wmma_f32_16x16x32_bf16 accumulators per wave).
// Tiles staged with double-buffered GLOBAL_LOAD_ASYNC_TO_LDS_B128:
// copies for tile k+1 overlap WMMA on tile k, gated by s_wait_asynccnt.
// EPI==0: f32 out.  EPI==1: tanh(v)*2 f32 out + sum(v^2) into *red.
// EPI==2: bf16 out (feeds the next GEMM's async A path directly).
// ---------------------------------------------------------------------------
#define TILE_BYTES (128 * 80)    // 128 rows * 40 ushorts (stride 80B, 16B aligned)

template <int EPI>
__global__ __launch_bounds__(256) void wmma_gemm(
    const u16* __restrict__ A, const u16* __restrict__ Wt,
    const float* __restrict__ bias, void* __restrict__ Cout,
    int M, int N, int Kd, float* __restrict__ red)
{
  __shared__ __align__(16) u16 As[2][128][40];  // bf16 [m][k], double buffered
  __shared__ __align__(16) u16 Bs[2][128][40];  // bf16 [n][k], double buffered

  const int t    = threadIdx.x;
  const int lane = t & 31;
  const int wv   = t >> 5;        // wave id 0..7
  const int wm   = wv & 3;        // 4 waves along M (32 rows each)
  const int wn   = wv >> 2;       // 2 waves along N (64 cols each)
  const int bm   = blockIdx.y * 128;
  const int bn   = blockIdx.x * 128;
  const int r16  = lane & 15;
  const int hi   = lane >> 4;

  const unsigned asBase = (unsigned)(uintptr_t)&As[0][0][0];
  const unsigned bsBase = (unsigned)(uintptr_t)&Bs[0][0][0];

  // Each thread owns 2 16-byte chunks of the A tile and 2 of the B tile.
  // chunk c (0..511): row = c>>2, k-chunk = c&3 (8 bf16 = 16B along K).
  const int ch0  = t * 2;
  const int row0 = ch0 >> 2,      kc0 = ch0 & 3;
  const int row1 = (ch0 + 1) >> 2, kc1 = (ch0 + 1) & 3;

  auto issue_tile = [&](int buf, int k0) {
    const unsigned lb = (unsigned)(buf * TILE_BYTES);
    async_b128(asBase + lb + row0 * 80 + kc0 * 16, A,
               (unsigned)(((bm + row0) * Kd + k0 + kc0 * 8) * 2));
    async_b128(asBase + lb + row1 * 80 + kc1 * 16, A,
               (unsigned)(((bm + row1) * Kd + k0 + kc1 * 8) * 2));
    async_b128(bsBase + lb + row0 * 80 + kc0 * 16, Wt,
               (unsigned)(((bn + row0) * Kd + k0 + kc0 * 8) * 2));
    async_b128(bsBase + lb + row1 * 80 + kc1 * 16, Wt,
               (unsigned)(((bn + row1) * Kd + k0 + kc1 * 8) * 2));
  };

  v8f acc[2][4];
#pragma unroll
  for (int i = 0; i < 2; i++)
#pragma unroll
    for (int j = 0; j < 4; j++)
#pragma unroll
      for (int v = 0; v < 8; v++) acc[i][j][v] = 0.0f;

  const int NT = Kd >> 5;          // K tiles of 32
  issue_tile(0, 0);                // prologue: 4 async ops in flight

  for (int kt = 0; kt < NT; ++kt) {
    const int cur = kt & 1;
    if (kt + 1 < NT) {
      issue_tile(cur ^ 1, (kt + 1) << 5);       // overlap next tile's copies
      asm volatile("s_wait_asynccnt 4" ::: "memory");  // tile kt resident
    } else {
      asm volatile("s_wait_asynccnt 0" ::: "memory");
    }
    __syncthreads();   // all waves' copies for tile kt visible

    // Fragments per CDNA5 16-bit WMMA VGPR layouts (ISA 7.12.2)
    union { unsigned u[8]; v16bf v; } af[2], bf4[4];
#pragma unroll
    for (int i = 0; i < 2; i++) {
      const int r = wm * 32 + i * 16 + r16;
#pragma unroll
      for (int j = 0; j < 8; j++) {
        const int kk = (hi << 3) + ((j >> 2) << 4) + ((j & 3) << 1);
        af[i].u[j] = *(const unsigned*)&As[cur][r][kk];
      }
    }
#pragma unroll
    for (int j = 0; j < 4; j++) {
      const int c = wn * 64 + j * 16 + r16;
#pragma unroll
      for (int q = 0; q < 8; q++) {
        bf4[j].u[q] = *(const unsigned*)&Bs[cur][c][(hi << 3) + ((q >> 2) << 4) + ((q & 3) << 1)];
      }
    }
#pragma unroll
    for (int i = 0; i < 2; i++)
#pragma unroll
      for (int j = 0; j < 4; j++)
        acc[i][j] = __builtin_amdgcn_wmma_f32_16x16x32_bf16(
            false, af[i].v, false, bf4[j].v, (short)0, acc[i][j], false, false);

    __syncthreads();   // reads done before buffer `cur` is refilled (kt+2)
  }

  // epilogue
  float rsum = 0.0f;
#pragma unroll
  for (int i = 0; i < 2; i++) {
#pragma unroll
    for (int j = 0; j < 4; j++) {
      const int col = bn + wn * 64 + j * 16 + r16;
      const float bb = bias[col];
#pragma unroll
      for (int v = 0; v < 8; v++) {
        const int row = bm + wm * 32 + i * 16 + hi * 8 + v;
        float val = acc[i][j][v] + bb;
        if (EPI == 1) { val = tanhf(val) * 2.0f; rsum += val * val; }
        if (EPI == 2) {
          ((u16*)Cout)[(size_t)row * N + col] = f2bf(val);
        } else {
          ((float*)Cout)[(size_t)row * N + col] = val;
        }
      }
    }
  }
  if (EPI == 1) {
    __shared__ float rs[256];
    rs[t] = rsum;
    __syncthreads();
    for (int s = 128; s > 0; s >>= 1) {
      if (t < s) rs[t] += rs[t + s];
      __syncthreads();
    }
    if (t == 0) atomicAdd(red, rs[0]);
  }
}

// ---------------------------------------------------------------------------
// fp32 -> bf16 convert (for x, the only activation not produced on-device)
// ---------------------------------------------------------------------------
__global__ __launch_bounds__(256) void convert_bf16_kernel(
    const float* __restrict__ src, u16* __restrict__ dst)
{
  const size_t i = (size_t)blockIdx.x * 256 + threadIdx.x;
  dst[i] = f2bf(src[i]);
}

// ---------------------------------------------------------------------------
// Weight transpose + convert: W (K,N) fp32 -> Wt (N,K) bf16, done once so
// B tiles are contiguous along K and can be staged with async B128 copies.
// ---------------------------------------------------------------------------
__global__ __launch_bounds__(256) void transpose_bf16_kernel(
    const float* __restrict__ W, u16* __restrict__ Wt, int N, int Kd)
{
  const size_t i = (size_t)blockIdx.x * 256 + threadIdx.x;  // over N*Kd
  const int k = (int)(i % Kd);
  const int n = (int)(i / Kd);
  Wt[i] = f2bf(W[(size_t)k * N + n]);
}

// ---------------------------------------------------------------------------
// Depthwise 3-tap conv (zero pad) + bias + SiLU -> bf16 (GEMM-A ready)
// ---------------------------------------------------------------------------
__global__ __launch_bounds__(256) void dwconv_silu_kernel(
    const float* __restrict__ x, const float* __restrict__ w,
    const float* __restrict__ b, u16* __restrict__ h)
{
  const size_t i = (size_t)blockIdx.x * 256 + threadIdx.x;
  const int c = (int)(i & (C_ - 1));
  const size_t bl = i >> 8;               // C_ == 256
  const int l = (int)(bl & (L_ - 1));
  const float xm = (l > 0)      ? x[i - C_] : 0.0f;
  const float x0 = x[i];
  const float xp = (l < L_ - 1) ? x[i + C_] : 0.0f;
  const float v = xm * w[c * 3 + 0] + x0 * w[c * 3 + 1] + xp * w[c * 3 + 2] + b[c];
  h[i] = f2bf(siluf(v));
}

// ---------------------------------------------------------------------------
// Scalars + tiny kernel-net MLP: sigma/env (thread 0), kernel_weights (K x C),
// and zero the two reduction accumulators.
// ---------------------------------------------------------------------------
__global__ __launch_bounds__(256) void setup_kernel(
    const float* __restrict__ raw_sigma,
    const float* __restrict__ k0w, const float* __restrict__ k0b,
    const float* __restrict__ k1w, const float* __restrict__ k1b,
    const float* __restrict__ k2w, const float* __restrict__ k2b,
    const float* __restrict__ k3w, const float* __restrict__ k3b,
    float* __restrict__ env, float* __restrict__ kw, float* __restrict__ scal)
{
  const int j = threadIdx.x;
  if (j < 2) scal[j] = 0.0f;
  if (j == 0) {
    float s = log1pf(__expf(raw_sigma[0]));        // softplus
    s = fminf(fmaxf(s, 0.05f), 0.5f);
    float e[K_]; float sum = 0.0f;
    for (int k = 0; k < K_; k++) {
      const float g = -0.5f + (float)k * (1.0f / 6.0f);
      const float d = g / s;
      e[k] = __expf(-0.5f * d * d);
      sum += e[k];
    }
    sum = fmaxf(sum, 1e-8f);
    for (int k = 0; k < K_; k++) env[k] = e[k] / sum;
  }
  // kernel net: Linear(1,32)->SiLU x3 -> Linear(32,256); thread j = out column
  float h0[HID_], h1[HID_], h2[HID_];
  for (int k = 0; k < K_; k++) {
    const float gin = (-0.5f + (float)k * (1.0f / 6.0f)) * 30.0f;  // OMEGA=30
    for (int o = 0; o < HID_; o++) h0[o] = siluf(gin * k0w[o] + k0b[o]);
    for (int o = 0; o < HID_; o++) {
      float a = k1b[o];
      for (int i2 = 0; i2 < HID_; i2++) a += h0[i2] * k1w[i2 * HID_ + o];
      h1[o] = siluf(a);
    }
    for (int o = 0; o < HID_; o++) {
      float a = k2b[o];
      for (int i2 = 0; i2 < HID_; i2++) a += h1[i2] * k2w[i2 * HID_ + o];
      h2[o] = siluf(a);
    }
    float a = k3b[j];
    for (int i2 = 0; i2 < HID_; i2++) a += h2[i2] * k3w[i2 * C_ + j];
    kw[k * C_ + j] = a;   // row-major (K, G*GC); consumer uses raw reshape g*7+k
  }
}

// ---------------------------------------------------------------------------
// Fused: envelope-scaled softmax over K, linear-interp gather from x_proj,
// implicit-kernel weighting, attention-weighted sum, entropy reduction.
// One block per (b,l); thread = channel (= group, GC=1). Output bf16 (GEMM A).
// ---------------------------------------------------------------------------
__global__ __launch_bounds__(256) void sample_combine_kernel(
    const float* __restrict__ xproj, const float* __restrict__ offs,
    const float* __restrict__ maskb, const float* __restrict__ env,
    const float* __restrict__ kw, u16* __restrict__ outpre,
    float* __restrict__ ent_acc)
{
  const int bl = blockIdx.x;
  const int c  = threadIdx.x;
  const int b  = bl >> 12;            // L_ == 4096
  const int l  = bl & (L_ - 1);
  const float* xb = xproj + (size_t)b * L_ * C_;
  const size_t base = (size_t)bl * NOFF + (size_t)c * K_;

  float m[K_];
  float mmax = -3.0e38f;
#pragma unroll
  for (int k = 0; k < K_; k++) {
    const float mv = maskb[base + k] * env[k];
    m[k] = mv;
    mmax = fmaxf(mmax, mv);
  }
  float s = 0.0f;
#pragma unroll
  for (int k = 0; k < K_; k++) { m[k] = __expf(m[k] - mmax); s += m[k]; }
  const float inv = 1.0f / s;

  float acc = 0.0f, entl = 0.0f;
#pragma unroll
  for (int k = 0; k < K_; k++) {
    const float at = m[k] * inv;
    entl += at * logf(at + 1e-8f);
    const float off = offs[base + k];                 // already tanh*2
    float p = (float)l + (float)(k - 3) + off;        // ref_off = k - K//2
    p = fminf(fmaxf(p, 0.0f), (float)(L_ - 1));
    const int pf = (int)p;                             // p >= 0 -> trunc == floor
    const int pc = min(pf + 1, L_ - 1);
    const float wc = p - (float)pf;
    const float samp = xb[(size_t)pf * C_ + c] * (1.0f - wc) +
                       xb[(size_t)pc * C_ + c] * wc;
    acc += samp * kw[c * K_ + k] * at;                 // raw-reshape kw[g*7+k]
  }
  outpre[(size_t)bl * C_ + c] = f2bf(acc);

  __shared__ float rs[256];
  rs[c] = entl;
  __syncthreads();
  for (int sft = 128; sft > 0; sft >>= 1) {
    if (c < sft) rs[c] += rs[c + sft];
    __syncthreads();
  }
  if (c == 0) atomicAdd(ent_acc, rs[0]);
}

__global__ void finalize_kernel(const float* __restrict__ scal,
                                float* __restrict__ out_tail)
{
  // offset_reg = mean over B*L*G*K*NDIM; -entropy = mean over B*L*G of sum_k
  out_tail[0] = scal[0] / 29360128.0f;   // 4*4096*256*7
  out_tail[1] = scal[1] / 4194304.0f;    // 4*4096*256
}

// ---------------------------------------------------------------------------
extern "C" void kernel_launch(void* const* d_in, const int* in_sizes, int n_in,
                              void* d_out, int out_size, void* d_ws, size_t ws_size,
                              hipStream_t stream)
{
  const float* x       = (const float*)d_in[0];
  const float* raw_sig = (const float*)d_in[1];
  const float* w_in    = (const float*)d_in[2];
  const float* b_in    = (const float*)d_in[3];
  const float* w_out   = (const float*)d_in[4];
  const float* b_out   = (const float*)d_in[5];
  const float* dw1_w   = (const float*)d_in[6];
  const float* dw1_b   = (const float*)d_in[7];
  const float* dw2_w   = (const float*)d_in[8];
  const float* dw2_b   = (const float*)d_in[9];
  const float* w_off   = (const float*)d_in[10];
  const float* b_off   = (const float*)d_in[11];
  const float* w_mask  = (const float*)d_in[12];
  const float* b_mask  = (const float*)d_in[13];
  const float* k0w     = (const float*)d_in[14];
  const float* k0b     = (const float*)d_in[15];
  const float* k1w     = (const float*)d_in[16];
  const float* k1b     = (const float*)d_in[17];
  const float* k2w     = (const float*)d_in[18];
  const float* k2b     = (const float*)d_in[19];
  const float* k3w     = (const float*)d_in[20];
  const float* k3b     = (const float*)d_in[21];

  // workspace carve (256B aligned slices)
  char* p = (char*)d_ws;
  auto carve = [&](size_t bytes) {
    char* r = p;
    p += (bytes + 255) & ~(size_t)255;
    return r;
  };
  float* xproj   = (float*)carve((size_t)M_ * C_ * 4);
  float* offs    = (float*)carve((size_t)M_ * NOFF * 4);
  float* maskb   = (float*)carve((size_t)M_ * NOFF * 4);
  u16*   xbf     = (u16*)  carve((size_t)M_ * C_ * 2);
  u16*   hbf     = (u16*)  carve((size_t)M_ * C_ * 2);
  u16*   xdwbf   = (u16*)  carve((size_t)M_ * C_ * 2);
  u16*   outpre  = (u16*)  carve((size_t)M_ * C_ * 2);
  u16*   wt_in   = (u16*)  carve((size_t)C_ * C_ * 2);
  u16*   wt_dw2  = (u16*)  carve((size_t)C_ * C_ * 2);
  u16*   wt_out  = (u16*)  carve((size_t)C_ * C_ * 2);
  u16*   wt_off  = (u16*)  carve((size_t)NOFF * C_ * 2);
  u16*   wt_mask = (u16*)  carve((size_t)NOFF * C_ * 2);
  float* env     = (float*)carve(32);
  float* kw      = (float*)carve((size_t)NOFF * 4);
  float* scal    = (float*)carve(8);

  float* outm = (float*)d_out;
  const dim3 blk(256);

  setup_kernel<<<1, blk, 0, stream>>>(raw_sig, k0w, k0b, k1w, k1b, k2w, k2b,
                                      k3w, k3b, env, kw, scal);
  // one-time bf16 conversions (weights transposed for async-K-contiguous tiles)
  convert_bf16_kernel<<<dim3(M_ * C_ / 256), blk, 0, stream>>>(x, xbf);
  transpose_bf16_kernel<<<dim3(C_ * C_ / 256), blk, 0, stream>>>(w_in,   wt_in,   C_,   C_);
  transpose_bf16_kernel<<<dim3(C_ * C_ / 256), blk, 0, stream>>>(dw2_w,  wt_dw2,  C_,   C_);
  transpose_bf16_kernel<<<dim3(C_ * C_ / 256), blk, 0, stream>>>(w_out,  wt_out,  C_,   C_);
  transpose_bf16_kernel<<<dim3(NOFF * C_ / 256), blk, 0, stream>>>(w_off,  wt_off,  NOFF, C_);
  transpose_bf16_kernel<<<dim3(NOFF * C_ / 256), blk, 0, stream>>>(w_mask, wt_mask, NOFF, C_);

  wmma_gemm<0><<<dim3(C_ / 128, M_ / 128), blk, 0, stream>>>(
      xbf, wt_in, b_in, xproj, M_, C_, C_, nullptr);
  dwconv_silu_kernel<<<dim3(M_), blk, 0, stream>>>(x, dw1_w, dw1_b, hbf);
  wmma_gemm<2><<<dim3(C_ / 128, M_ / 128), blk, 0, stream>>>(
      hbf, wt_dw2, dw2_b, xdwbf, M_, C_, C_, nullptr);
  wmma_gemm<1><<<dim3(NOFF / 128, M_ / 128), blk, 0, stream>>>(
      xdwbf, wt_off, b_off, offs, M_, NOFF, C_, scal + 0);
  wmma_gemm<0><<<dim3(NOFF / 128, M_ / 128), blk, 0, stream>>>(
      xdwbf, wt_mask, b_mask, maskb, M_, NOFF, C_, nullptr);
  sample_combine_kernel<<<dim3(M_), blk, 0, stream>>>(
      xproj, offs, maskb, env, kw, outpre, scal + 1);
  wmma_gemm<0><<<dim3(C_ / 128, M_ / 128), blk, 0, stream>>>(
      outpre, wt_out, b_out, outm, M_, C_, C_, nullptr);
  finalize_kernel<<<1, 1, 0, stream>>>(scal, outm + (size_t)M_ * C_);
}
